// Attention_3401614098861
// MI455X (gfx1250) — matrix operand order repaired and checked by reading, not compiled, not run on hardware
//
#include <hip/hip_runtime.h>
#include <math.h>
#include <stdint.h>

// Problem constants (from the reference): H=1024, B=16, S=1024
#define H_DIM   1024
#define B_DIM   16
#define S_DIM   1024
#define K_TOT   2048            // 2H, reduction dim of e_proj GEMM
#define M_TOT   16384           // S*B rows
#define W_STRIDE 3072           // attn_w row length (3H)

// GEMM tiling
#define BM 64
#define BN 128
#define BK 32
#define LDA 40                  // padded LDS row stride (elements) to avoid bank conflicts
#define KSTEPS (K_TOT / BK)     // 64

typedef __attribute__((ext_vector_type(16))) __bf16 v16bf;
typedef __attribute__((ext_vector_type(8)))  float  v8f;

union Frag { v16bf v; uint4 u[2]; };

__device__ __forceinline__ unsigned short f2bf_rne(float f) {
    unsigned int x = __float_as_uint(f);
    x += 0x7FFFu + ((x >> 16) & 1u);      // round-to-nearest-even
    return (unsigned short)(x >> 16);
}
__device__ __forceinline__ float bf2f(unsigned short h) {
    return __uint_as_float(((unsigned int)h) << 16);
}

// ---------------------------------------------------------------------------
// Pack w_e = attn_w[:, H:3H] into bf16 hi/lo, fragment-ready tiling:
//   dst[((tile_o*64 + kc) << 9) + (o%16)*32 + (k%32)]
// so a wave's B fragment (lane lr = column, K chunk of 16) is a contiguous,
// fully coalesced 1 KB block per 16x32 tile.
// ---------------------------------------------------------------------------
__global__ void pack_we_kernel(const float* __restrict__ w,
                               unsigned short* __restrict__ bhi,
                               unsigned short* __restrict__ blo) {
    int g = blockIdx.x * blockDim.x + threadIdx.x;   // 0 .. H*K_TOT-1
    int o = g >> 11;          // 0..1023
    int k = g & 2047;         // 0..2047
    float f = w[(size_t)o * W_STRIDE + H_DIM + k];
    unsigned short hi = f2bf_rne(f);
    unsigned short lo = f2bf_rne(f - bf2f(hi));
    int tile_o = o >> 4, n = o & 15, kc = k >> 5, kl = k & 31;
    size_t dst = ((size_t)(tile_o * 64 + kc) << 9) + (n << 5) + kl;
    bhi[dst] = hi;
    blo[dst] = lo;
}

// hb[b*H + o] = attn_b[o] + sum_h hidden[b,h] * attn_w[o, h]   (w_h = attn_w[:, :H])
__global__ void hproj_kernel(const float* __restrict__ hidden,
                             const float* __restrict__ w,
                             const float* __restrict__ bias,
                             float* __restrict__ hb) {
    int g = blockIdx.x * blockDim.x + threadIdx.x;   // 0 .. 16383
    int o = g >> 4;
    int b = g & 15;
    const float* wr = w + (size_t)o * W_STRIDE;
    const float* hr = hidden + (size_t)b * H_DIM;
    float acc = bias[o];
#pragma unroll 8
    for (int h = 0; h < H_DIM; ++h) acc = fmaf(hr[h], wr[h], acc);
    hb[b * H_DIM + o] = acc;
}

// ---------------------------------------------------------------------------
// Fused: e_proj GEMM (bf16x3 WMMA, f32 accumulate) + hb add + tanh + dot(v).
// A tiles: async-copied f32 -> LDS (global_load_async_to_lds_b128, ASYNCcnt
//          pipelined one K-step ahead), split hi/lo bf16 in LDS, ds_load frags.
// B tiles: prepacked bf16 hi/lo, fragment-ready in global (L2-resident),
//          loaded straight into registers.
// ---------------------------------------------------------------------------
__global__ __launch_bounds__(256) void energy_gemm_kernel(
        const float* __restrict__ enc,            // (S*B, 2H), m = s*B + b
        const unsigned short* __restrict__ bhi,   // packed w_e hi
        const unsigned short* __restrict__ blo,   // packed w_e lo
        const float* __restrict__ vvec,           // (H)
        const float* __restrict__ hb,             // (B, H)
        float* __restrict__ scores) {             // (M_TOT) atomically accumulated
    __shared__ __align__(16) float          sF[2][BM * BK];        // raw f32 A tiles
    __shared__ __align__(16) unsigned short sA[2][2][BM * LDA];    // [buf][hi/lo]

    const int tid  = threadIdx.x;
    const int lane = tid & 31;
    const int wave = tid >> 5;
    const int wm   = wave >> 2;          // 0..1 (M direction)
    const int wn   = wave & 3;           // 0..3 (N direction)
    const int lr   = lane & 15;
    const int half = lane >> 4;
    const int mBase = blockIdx.y * BM;

    // per-thread A-copy assignment: 16B chunk q covers row r = q>>3, cols (q&7)*4..+3
    const int q0 = tid, q1 = tid + 256;
    const int r0 = q0 >> 3, c0 = q0 & 7;
    const int r1 = q1 >> 3, c1 = q1 & 7;
    const unsigned long long ga0 =
        (unsigned long long)(uintptr_t)enc + ((size_t)(mBase + r0) * K_TOT + c0 * 4) * 4ull;
    const unsigned long long ga1 =
        (unsigned long long)(uintptr_t)enc + ((size_t)(mBase + r1) * K_TOT + c1 * 4) * 4ull;
    const unsigned sFb0 = (unsigned)(uintptr_t)&sF[0][0];   // LDS byte offsets
    const unsigned sFb1 = (unsigned)(uintptr_t)&sF[1][0];

    v8f acc[2][2] = {};                  // f32 accumulators, 2x2 16x16 D tiles per wave

    // prologue: async-copy K-step 0 into sF[0]
    {
        unsigned d0 = sFb0 + q0 * 16, d1 = sFb0 + q1 * 16;
        asm volatile("global_load_async_to_lds_b128 %0, %1, off"
                     :: "v"(d0), "v"(ga0) : "memory");
        asm volatile("global_load_async_to_lds_b128 %0, %1, off"
                     :: "v"(d1), "v"(ga1) : "memory");
    }

    for (int step = 0; step < KSTEPS; ++step) {
        const int buf = step & 1;

        // kick off async copies for the next K-step, then retire this one's
        if (step + 1 < KSTEPS) {
            unsigned long long koff = (unsigned long long)(step + 1) * (BK * 4);
            unsigned base = (buf ^ 1) ? sFb1 : sFb0;
            unsigned d0 = base + q0 * 16, d1 = base + q1 * 16;
            asm volatile("global_load_async_to_lds_b128 %0, %1, off"
                         :: "v"(d0), "v"(ga0 + koff) : "memory");
            asm volatile("global_load_async_to_lds_b128 %0, %1, off"
                         :: "v"(d1), "v"(ga1 + koff) : "memory");
            asm volatile("s_wait_asynccnt 0x2" ::: "memory");   // older 2 done (in-order)
        } else {
            asm volatile("s_wait_asynccnt 0x0" ::: "memory");
        }
        __syncthreads();   // every thread retired its own copies -> tile visible

        // split f32 tile -> bf16 hi/lo in LDS (each element converted once)
#pragma unroll
        for (int i = 0; i < 2; ++i) {
            const int q = i ? q1 : q0, r = i ? r1 : r0, c = i ? c1 : c0;
            float4 fv4 = *reinterpret_cast<const float4*>(&sF[buf][q * 4]);
            float f[4] = { fv4.x, fv4.y, fv4.z, fv4.w };
            unsigned short hh[4], ll[4];
#pragma unroll
            for (int j = 0; j < 4; ++j) {
                hh[j] = f2bf_rne(f[j]);
                ll[j] = f2bf_rne(f[j] - bf2f(hh[j]));
            }
            uint2 hv = make_uint2((unsigned)hh[0] | ((unsigned)hh[1] << 16),
                                  (unsigned)hh[2] | ((unsigned)hh[3] << 16));
            uint2 lv = make_uint2((unsigned)ll[0] | ((unsigned)ll[1] << 16),
                                  (unsigned)ll[2] | ((unsigned)ll[3] << 16));
            *reinterpret_cast<uint2*>(&sA[buf][0][r * LDA + c * 4]) = hv;
            *reinterpret_cast<uint2*>(&sA[buf][1][r * LDA + c * 4]) = lv;
        }
        __syncthreads();

        // A fragments from LDS (documented 16-bit A striping, wave32)
        Frag ah[2], al[2], bh[2], bl[2];
#pragma unroll
        for (int tm = 0; tm < 2; ++tm) {
            const unsigned short* p0 = &sA[buf][0][(wm * 32 + tm * 16 + lr) * LDA + half * 8];
            const unsigned short* p1 = &sA[buf][1][(wm * 32 + tm * 16 + lr) * LDA + half * 8];
            ah[tm].u[0] = *reinterpret_cast<const uint4*>(p0);        // K = 8h..8h+7
            ah[tm].u[1] = *reinterpret_cast<const uint4*>(p0 + 16);   // K = 16+8h..16+8h+7
            al[tm].u[0] = *reinterpret_cast<const uint4*>(p1);
            al[tm].u[1] = *reinterpret_cast<const uint4*>(p1 + 16);
        }
        // B fragments straight from packed global (L2-resident, coalesced)
#pragma unroll
        for (int tn = 0; tn < 2; ++tn) {
            int tileO = blockIdx.x * 8 + wn * 2 + tn;
            size_t base = ((size_t)(tileO * 64 + step) << 9) + (lr << 5) + (half << 3);
            bh[tn].u[0] = *reinterpret_cast<const uint4*>(bhi + base);     // K = 8h..8h+7, 16+8h..16+8h+7
            bh[tn].u[1] = *reinterpret_cast<const uint4*>(bhi + base + 16);
            bl[tn].u[0] = *reinterpret_cast<const uint4*>(blo + base);
            bl[tn].u[1] = *reinterpret_cast<const uint4*>(blo + base + 16);
        }

        // bf16x3: hi*hi + hi*lo + lo*hi, all into f32 accumulators
#pragma unroll
        for (int tm = 0; tm < 2; ++tm)
#pragma unroll
            for (int tn = 0; tn < 2; ++tn) {
                acc[tm][tn] = __builtin_amdgcn_wmma_f32_16x16x32_bf16(
                    false, ah[tm].v, false, bh[tn].v, (short)0, acc[tm][tn], false, false);
                acc[tm][tn] = __builtin_amdgcn_wmma_f32_16x16x32_bf16(
                    false, ah[tm].v, false, bl[tn].v, (short)0, acc[tm][tn], false, false);
                acc[tm][tn] = __builtin_amdgcn_wmma_f32_16x16x32_bf16(
                    false, al[tm].v, false, bh[tn].v, (short)0, acc[tm][tn], false, false);
            }
    }

    // epilogue: scores[m] += sum_o tanh(acc + hb[b,o]) * v[o]
    // m = s*B + b and BM is a multiple of B, so b = (i + 8*half), independent of tm.
    const int oBase = blockIdx.x * BN;
    float vv[2], hbv[2][8];
#pragma unroll
    for (int tn = 0; tn < 2; ++tn) {
        int o = oBase + wn * 32 + tn * 16 + lr;
        vv[tn] = vvec[o];
#pragma unroll
        for (int i = 0; i < 8; ++i)
            hbv[tn][i] = hb[(i + 8 * half) * H_DIM + o];
    }
#pragma unroll
    for (int tm = 0; tm < 2; ++tm) {
#pragma unroll
        for (int i = 0; i < 8; ++i) {
            float val = tanhf(acc[tm][0][i] + hbv[0][i]) * vv[0]
                      + tanhf(acc[tm][1][i] + hbv[1][i]) * vv[1];
#pragma unroll
            for (int off = 1; off < 16; off <<= 1)
                val += __shfl_xor(val, off, 16);
            if (lr == 0) {
                int r = wm * 32 + tm * 16 + i + 8 * half;
                atomicAdd(&scores[mBase + r], val);
            }
        }
    }
}

// softmax over s for each b; scores index m = s*B + b, output is (B, S)
__global__ void softmax_kernel(const float* __restrict__ scores,
                               float* __restrict__ out) {
    const int b = blockIdx.x;
    const int t = threadIdx.x;
    __shared__ float red[256];

    float m = -INFINITY;
    float x[4];
#pragma unroll
    for (int i = 0; i < 4; ++i) {
        int s = t + i * 256;
        x[i] = scores[s * B_DIM + b];
        m = fmaxf(m, x[i]);
    }
    red[t] = m; __syncthreads();
    for (int off = 128; off > 0; off >>= 1) {
        if (t < off) red[t] = fmaxf(red[t], red[t + off]);
        __syncthreads();
    }
    float mx = red[0]; __syncthreads();

    float sum = 0.0f;
#pragma unroll
    for (int i = 0; i < 4; ++i) { x[i] = expf(x[i] - mx); sum += x[i]; }
    red[t] = sum; __syncthreads();
    for (int off = 128; off > 0; off >>= 1) {
        if (t < off) red[t] += red[t + off];
        __syncthreads();
    }
    float inv = 1.0f / red[0];
#pragma unroll
    for (int i = 0; i < 4; ++i) {
        int s = t + i * 256;
        out[(size_t)b * S_DIM + s] = x[i] * inv;
    }
}

extern "C" void kernel_launch(void* const* d_in, const int* in_sizes, int n_in,
                              void* d_out, int out_size, void* d_ws, size_t ws_size,
                              hipStream_t stream) {
    const float* hidden = (const float*)d_in[0];   // (B, H)
    const float* enc    = (const float*)d_in[1];   // (S, B, 2H)
    const float* attn_w = (const float*)d_in[2];   // (H, 3H)
    const float* attn_b = (const float*)d_in[3];   // (H)
    const float* vvec   = (const float*)d_in[4];   // (H)
    float* out = (float*)d_out;                    // (B, S)

    float*          scores = (float*)d_ws;                          // 64 KB
    float*          hb     = scores + M_TOT;                        // 64 KB
    unsigned short* bhi    = (unsigned short*)(hb + B_DIM * H_DIM); // 4 MB
    unsigned short* blo    = bhi + (size_t)H_DIM * K_TOT;           // 4 MB

    (void)in_sizes; (void)n_in; (void)out_size; (void)ws_size;

    hipMemsetAsync(scores, 0, (size_t)M_TOT * sizeof(float), stream);
    pack_we_kernel<<<(H_DIM * K_TOT) / 256, 256, 0, stream>>>(attn_w, bhi, blo);
    hproj_kernel<<<M_TOT / 256, 256, 0, stream>>>(hidden, attn_w, attn_b, hb);
    energy_gemm_kernel<<<dim3(H_DIM / BN, M_TOT / BM), 256, 0, stream>>>(
        enc, bhi, blo, vvec, hb, scores);
    softmax_kernel<<<B_DIM, 256, 0, stream>>>(scores, out);
}
